// SpanClassifier_63900523429940
// MI455X (gfx1250) — compile-verified, matrix operand order checked by reading, not run on hardware
//
#include <hip/hip_runtime.h>

typedef __attribute__((ext_vector_type(16))) _Float16 v16h;
typedef __attribute__((ext_vector_type(8)))  _Float16 v8h;
typedef __attribute__((ext_vector_type(4)))  _Float16 v4h;
typedef __attribute__((ext_vector_type(8)))  float    v8f;
typedef __attribute__((ext_vector_type(4)))  float    v4f;

#define SEQ   2048
#define BATCH 8
#define DIM   256
#define RELP  144          // 2K+1 = 129 padded to 9 x 16 tiles
#define NEG_BIG -1e18f

// Load one 16x32 f16 WMMA operand fragment from a row-major [rows x stride]
// array whose 32-wide K window starts at `base` (K contiguous).
// Per CDNA5 ISA 7.12.2 (16-bit A 16x32): lanes 0-15 hold rows 0-15 with
// K=0..7 (VGPR0-3) and K=16..23 (VGPR4-7); lanes 16-31 hold K=8..15 / 24..31.
// B operand assumed symmetric (lane n = column n, same K packing), which makes
// every GEMM here an NT-GEMM with a single loader. Two b128 loads per lane.
__device__ __forceinline__ v16h load_frag(const _Float16* __restrict__ base,
                                          int stride, int lane) {
  int row = lane & 15;
  int kh  = lane >> 4;                       // 0 or 1
  const _Float16* p = base + (size_t)row * stride + kh * 8;
  v8h lo = *(const v8h*)(p);                 // K = kh*8 + 0..7
  v8h hi = *(const v8h*)(p + 16);            // K = 16 + kh*8 + 0..7
  v16h r;
#pragma unroll
  for (int i = 0; i < 8; ++i) { r[i] = lo[i]; r[8 + i] = hi[i]; }
  return r;
}

// ---------------- f32 -> f16 conversion helpers ----------------

__global__ void cvt_f32_to_f16(const float* __restrict__ s,
                               _Float16* __restrict__ d, int n4) {
  int i = blockIdx.x * blockDim.x + threadIdx.x;
  if (i < n4) {
    v4f v = *(const v4f*)(s + (size_t)i * 4);
    v4h h;
#pragma unroll
    for (int j = 0; j < 4; ++j) h[j] = (_Float16)v[j];
    *(v4h*)(d + (size_t)i * 4) = h;
  }
}

__global__ void rel_pad_f16(const float* __restrict__ s,
                            _Float16* __restrict__ d) {
  int i = blockIdx.x * blockDim.x + threadIdx.x;   // over RELP*DIM
  int r = i >> 8, c = i & 255;
  float v = (r < 129) ? s[r * DIM + c] : 0.0f;
  d[i] = (_Float16)v;
}

// ---------------- q/k projections: q=(x@WqT+bq)/16, k=x@WkT+bk ----------------

__global__ __launch_bounds__(128, 1) void proj_f16(
    const _Float16* __restrict__ xh,
    const _Float16* __restrict__ wqh, const float* __restrict__ bq,
    const _Float16* __restrict__ wkh, const float* __restrict__ bk,
    _Float16* __restrict__ qh, _Float16* __restrict__ kh_out) {
  int lane = threadIdx.x & 31;
  int wave = threadIdx.x >> 5;
  int m0 = (blockIdx.x * 4 + wave) * 16;           // row tile in [0, B*S)
  v16h a[8];
#pragma unroll
  for (int t = 0; t < 8; ++t)
    a[t] = load_frag(xh + (size_t)m0 * DIM + t * 32, DIM, lane);
  int n = lane & 15, hf = lane >> 4;
#pragma unroll 1
  for (int n0 = 0; n0 < DIM; n0 += 16) {           // q sweep
    v16h b[8];
#pragma unroll
    for (int t = 0; t < 8; ++t)
      b[t] = load_frag(wqh + (size_t)n0 * DIM + t * 32, DIM, lane);
    v8f c = {};
#pragma unroll
    for (int t = 0; t < 8; ++t)
      c = __builtin_amdgcn_wmma_f32_16x16x32_f16(false, a[t], false, b[t],
                                                 (short)0, c, false, false);
    float bias = bq[n0 + n];
#pragma unroll
    for (int v = 0; v < 8; ++v)                    // C: M = v + 8*hf, N = n
      qh[(size_t)(m0 + v + 8 * hf) * DIM + n0 + n] =
          (_Float16)((c[v] + bias) * 0.0625f);     // fold 1/sqrt(256)
  }
#pragma unroll 1
  for (int n0 = 0; n0 < DIM; n0 += 16) {           // k sweep, reuses A frags
    v16h b[8];
#pragma unroll
    for (int t = 0; t < 8; ++t)
      b[t] = load_frag(wkh + (size_t)n0 * DIM + t * 32, DIM, lane);
    v8f c = {};
#pragma unroll
    for (int t = 0; t < 8; ++t)
      c = __builtin_amdgcn_wmma_f32_16x16x32_f16(false, a[t], false, b[t],
                                                 (short)0, c, false, false);
    float bias = bk[n0 + n];
#pragma unroll
    for (int v = 0; v < 8; ++v)
      kh_out[(size_t)(m0 + v + 8 * hf) * DIM + n0 + n] =
          (_Float16)(c[v] + bias);
  }
}

// ---------------- q_rel = q @ rel^T  (f32 table, [B*S x RELP]) ----------------

__global__ __launch_bounds__(128, 1) void qrel_f32(
    const _Float16* __restrict__ qh, const _Float16* __restrict__ relh,
    float* __restrict__ qrel) {
  int lane = threadIdx.x & 31;
  int wave = threadIdx.x >> 5;
  int m0 = (blockIdx.x * 4 + wave) * 16;
  v16h a[8];
#pragma unroll
  for (int t = 0; t < 8; ++t)
    a[t] = load_frag(qh + (size_t)m0 * DIM + t * 32, DIM, lane);
  int n = lane & 15, hf = lane >> 4;
#pragma unroll 1
  for (int n0 = 0; n0 < RELP; n0 += 16) {
    v16h b[8];
#pragma unroll
    for (int t = 0; t < 8; ++t)
      b[t] = load_frag(relh + (size_t)n0 * DIM + t * 32, DIM, lane);
    v8f c = {};
#pragma unroll
    for (int t = 0; t < 8; ++t)
      c = __builtin_amdgcn_wmma_f32_16x16x32_f16(false, a[t], false, b[t],
                                                 (short)0, c, false, false);
#pragma unroll
    for (int v = 0; v < 8; ++v)
      qrel[(size_t)(m0 + v + 8 * hf) * RELP + n0 + n] = c[v];
  }
}

// ---------------- scores = q@k^T + gather(q_rel) with mask ----------------
// Block = 8 waves x 16 rows = 128 q-rows of one batch; each wave keeps its
// q A-frags (64 VGPRs) resident and streams all S/16 k tiles (shared across
// waves -> WGP$/L2 broadcast). All 16 B-frag loads of a tile are issued in
// one clause before the WMMA chain for full memory-level parallelism.
// Output stores are non-temporal: 268MB of streaming writes must not evict
// the L2-resident k (16MB) and q_rel (19MB) operands.

__global__ __launch_bounds__(256, 1) void scores_kernel(
    const _Float16* __restrict__ qh, const _Float16* __restrict__ kh,
    const float* __restrict__ qrel, const int* __restrict__ mask,
    float* __restrict__ out) {
  int lane = threadIdx.x & 31;
  int wave = threadIdx.x >> 5;
  int bi = blockIdx.y;
  int m0 = blockIdx.x * 128 + wave * 16;           // q row within sequence
  size_t row0 = (size_t)bi * SEQ + m0;             // row into [B*S] arrays
  v16h a[8];
#pragma unroll
  for (int t = 0; t < 8; ++t)
    a[t] = load_frag(qh + row0 * DIM + t * 32, DIM, lane);
  int n = lane & 15, hf = lane >> 4;
  const float* qrb = qrel + row0 * RELP;           // + (v+8*hf)*RELP per row
  int mq[8];
#pragma unroll
  for (int v = 0; v < 8; ++v)
    mq[v] = mask[row0 + v + 8 * hf];
  float* outb = out + ((size_t)bi * SEQ + m0) * SEQ;
#pragma unroll 1
  for (int n0 = 0; n0 < SEQ; n0 += 16) {
    const _Float16* kb = kh + ((size_t)bi * SEQ + n0) * DIM;
    v16h b[8];
#pragma unroll
    for (int t = 0; t < 8; ++t)                    // 16 b128 loads, one clause
      b[t] = load_frag(kb + t * 32, DIM, lane);
    v8f c = {};
#pragma unroll
    for (int t = 0; t < 8; ++t)                    // back-to-back WMMA chain
      c = __builtin_amdgcn_wmma_f32_16x16x32_f16(false, a[t], false, b[t],
                                                 (short)0, c, false, false);
    int ki = n0 + n;
    int mk = mask[(size_t)bi * SEQ + ki];
#pragma unroll
    for (int v = 0; v < 8; ++v) {
      int m = v + 8 * hf;                          // local q row
      int d = ki - (m0 + m);
      d = d < -64 ? -64 : (d > 64 ? 64 : d);       // clip(k-q, -K, K)
      float s = c[v] + qrb[(size_t)m * RELP + d + 64];
      __builtin_nontemporal_store((mq[v] && mk) ? s : NEG_BIG,
                                  &outb[(size_t)m * SEQ + ki]);
    }
  }
}

// ---------------- host ----------------

extern "C" void kernel_launch(void* const* d_in, const int* in_sizes, int n_in,
                              void* d_out, int out_size, void* d_ws,
                              size_t ws_size, hipStream_t stream) {
  (void)in_sizes; (void)n_in; (void)out_size; (void)ws_size;
  const float* repre  = (const float*)d_in[0];
  const int*   mask   = (const int*)d_in[1];
  const float* wq[2]  = {(const float*)d_in[2],  (const float*)d_in[7]};
  const float* bq[2]  = {(const float*)d_in[3],  (const float*)d_in[8]};
  const float* wk[2]  = {(const float*)d_in[4],  (const float*)d_in[9]};
  const float* bk[2]  = {(const float*)d_in[5],  (const float*)d_in[10]};
  const float* rel[2] = {(const float*)d_in[6],  (const float*)d_in[11]};
  float* out = (float*)d_out;

  const size_t ROWS = (size_t)BATCH * SEQ;         // 16384
  char* ws = (char*)d_ws;
  size_t off = 0;
  auto take = [&](size_t bytes) -> char* {
    char* p = ws + off;
    off = (off + bytes + 255) & ~(size_t)255;
    return p;
  };
  _Float16* xh = (_Float16*)take(ROWS * DIM * 2);
  _Float16 *qh[2], *khd[2], *wqh[2], *wkh[2], *relh[2];
  float* qrel[2];
  for (int h = 0; h < 2; ++h) {
    qh[h]   = (_Float16*)take(ROWS * DIM * 2);
    khd[h]  = (_Float16*)take(ROWS * DIM * 2);
    wqh[h]  = (_Float16*)take((size_t)DIM * DIM * 2);
    wkh[h]  = (_Float16*)take((size_t)DIM * DIM * 2);
    relh[h] = (_Float16*)take((size_t)RELP * DIM * 2);
    qrel[h] = (float*)take(ROWS * RELP * 4);
  }

  {                                                // precision downcasts
    int n4 = (int)(ROWS * DIM / 4);
    cvt_f32_to_f16<<<(n4 + 255) / 256, 256, 0, stream>>>(repre, xh, n4);
    int w4 = DIM * DIM / 4;
    for (int h = 0; h < 2; ++h) {
      cvt_f32_to_f16<<<(w4 + 255) / 256, 256, 0, stream>>>(wq[h], wqh[h], w4);
      cvt_f32_to_f16<<<(w4 + 255) / 256, 256, 0, stream>>>(wk[h], wkh[h], w4);
      rel_pad_f16<<<RELP * DIM / 256, 256, 0, stream>>>(rel[h], relh[h]);
    }
  }
  for (int h = 0; h < 2; ++h) {                    // projections + rel table
    proj_f16<<<dim3((unsigned)(ROWS / 64)), 128, 0, stream>>>(
        xh, wqh[h], bq[h], wkh[h], bk[h], qh[h], khd[h]);
    qrel_f32<<<dim3((unsigned)(ROWS / 64)), 128, 0, stream>>>(
        qh[h], relh[h], qrel[h]);
  }
  for (int h = 0; h < 2; ++h) {                    // S x S logits
    scores_kernel<<<dim3(SEQ / 128, BATCH), 256, 0, stream>>>(
        qh[h], khd[h], qrel[h], mask,
        out + (size_t)h * BATCH * SEQ * SEQ);
  }
}
